// RgSTA_10187662426592
// MI455X (gfx1250) — matrix operand order repaired and checked by reading, not compiled, not run on hardware
//
#include <hip/hip_runtime.h>
#include <hip/hip_bf16.h>

// MI455X (gfx1250) fused chunk-merge kernel.
//
// Roofline: GEMM feat = tokens @ W^T is 2*65536*1024*1024 = 137 GFLOP.
// Min HBM traffic ~320MB (~14us @ 23.3TB/s); bf16 WMMA (16x16x32, f32 acc)
// lands at ~27us compute -> near balance, so the GEMM runs in bf16 and
// everything else (bias, Gram/LN/softmax epilogue, token merge) in f32.
// feat never touches HBM: sim, LN stats and the score head are additive
// reductions over the feature axis, accumulated per-lane in registers and
// reduced once per wave with DPP row-shift adds (VALU, no LDS traffic).
//
// Workspace requirement: 2MB (bf16 W fragments) + 4KB (gws) + 8B scalars.

typedef __bf16 bf16;
typedef __attribute__((ext_vector_type(16))) __bf16 v16bf;
typedef __attribute__((ext_vector_type(8)))  __bf16 v8bf;
typedef __attribute__((ext_vector_type(8)))  float  v8f;

#define D_DIM   1024
#define T_DIM   4096
#define B_DIM   16
#define MT_TOK  64       // tokens per block (16 chunks of 4)
#define NCHUNK  16
#define AP      1032     // bf16 LDS row pitch (mult of 8 for b128 loads)
#define FP      1028     // f32 LDS row pitch (phase-3 token staging)
#define TAU_INV 10.0f
#define ALPHA_C 0.7f

// 16-lane inclusive row sum via DPP row_shr adds; lanes 15/31 hold the sum.
__device__ __forceinline__ float row_sum16(float x) {
  int t;
  t = __builtin_amdgcn_update_dpp(0, __builtin_bit_cast(int, x), 0x111, 0xf, 0xf, true);
  x += __builtin_bit_cast(float, t);
  t = __builtin_amdgcn_update_dpp(0, __builtin_bit_cast(int, x), 0x112, 0xf, 0xf, true);
  x += __builtin_bit_cast(float, t);
  t = __builtin_amdgcn_update_dpp(0, __builtin_bit_cast(int, x), 0x114, 0xf, 0xf, true);
  x += __builtin_bit_cast(float, t);
  t = __builtin_amdgcn_update_dpp(0, __builtin_bit_cast(int, x), 0x118, 0xf, 0xf, true);
  x += __builtin_bit_cast(float, t);
  return x;
}

// ---------------------------------------------------------------------------
// Prep 1: repack W_proj (f32 row-major [e][d]) into bf16 WMMA B-fragments.
// feat = tokens @ W^T  =>  B[k][n] = W[n][k].
// Fragment (kb = k/32, nb = n/16), stored as 32 lanes x 16 bf16 (32B/lane):
//   lane l<16 : column n = nb*16+l,    K = kb*32 + 0..15 (ascending)
//   lane l>=16: column n = nb*16+l-16, K = kb*32 + 16..31
__global__ __launch_bounds__(256) void repack_w(const float* __restrict__ W,
                                                bf16* __restrict__ wB) {
  int gid  = blockIdx.x * 256 + threadIdx.x;   // 65536 = 2048 frags * 32 lanes
  int frag = gid >> 5, lane = gid & 31;
  int nb = frag >> 5, kb = frag & 31;
  int n  = nb * 16 + (lane & 15);
  int k0 = kb * 32 + ((lane >> 4) << 4);
  const float* src = W + (size_t)n * D_DIM + k0;
  bf16* dst = wB + (size_t)frag * 512 + lane * 16;
#pragma unroll
  for (int i = 0; i < 16; ++i) dst[i] = (bf16)src[i];
}

// ---------------------------------------------------------------------------
// Prep 2: gws[e] = ln_g[e]*w_score[e]; scal = { sum(gws), dot(ln_b, w_score) }
__global__ __launch_bounds__(256) void prep_vec(const float* __restrict__ ln_g,
                                                const float* __restrict__ ln_b,
                                                const float* __restrict__ w_score,
                                                float* __restrict__ gws,
                                                float* __restrict__ scal) {
  __shared__ float r1[256], r2[256];
  int tid = threadIdx.x;
  float s1 = 0.f, s2 = 0.f;
  for (int e = tid; e < D_DIM; e += 256) {
    float g = ln_g[e] * w_score[e];
    gws[e] = g;
    s1 += g;
    s2 += ln_b[e] * w_score[e];
  }
  r1[tid] = s1; r2[tid] = s2;
  __syncthreads();
  for (int off = 128; off > 0; off >>= 1) {
    if (tid < off) { r1[tid] += r1[tid + off]; r2[tid] += r2[tid + off]; }
    __syncthreads();
  }
  if (tid == 0) { scal[0] = r1[0]; scal[1] = r2[0]; }
}

// ---------------------------------------------------------------------------
// Main fused kernel: 1024 blocks (16 batches x 64 chunk-tiles), 256 thr.
__global__ __launch_bounds__(256, 1) void fused_tome(
    const float* __restrict__ x, const float* __restrict__ b_proj,
    const bf16* __restrict__ wB, const float* __restrict__ gws,
    const float* __restrict__ scal, float* __restrict__ out) {
  extern __shared__ char smem[];
  bf16*  Ab   = (bf16*)smem;                              // 64 x AP bf16
  float* Fs   = (float*)smem;                             // phase-3 alias
  float* aux  = (float*)(smem + (size_t)MT_TOK * AP * 2); // past Ab
  float* part  = aux;                  // [8 waves][16 chunks][18]
  float* impP  = part + 8 * 16 * 18;   // 256
  float* impA  = impP + 256;           // 64 token sq-norms
  float* fq    = impA + 64;            // [16][18] reduced quantities
  float* coefA = fq + 16 * 18;         // [16][4] merge coefficients

  const int tid  = threadIdx.x;
  const int wave = tid >> 5;
  const int lane = tid & 31;
  const int b     = blockIdx.x >> 6;
  const int ctile = blockIdx.x & 63;
  const int c0 = ctile * NCHUNK;
  const int t0 = ctile * MT_TOK;

  // ---- phase 1: coalesced x tile -> bf16 LDS + raw token sq-norms --------
  // i = tid + 256*n  =>  t = i & 63 is constant per thread.
  float sq = 0.f;
  for (int i = tid; i < MT_TOK * D_DIM; i += 256) {
    int d = i >> 6, t = i & 63;
    float v = x[((size_t)b * D_DIM + d) * T_DIM + t0 + t];
    Ab[t * AP + d] = (bf16)v;
    sq += v * v;
  }
  impP[tid] = sq;
  __syncthreads();   // Ab complete, impP complete
  if (tid < 64)
    impA[tid] = impP[tid] + impP[tid + 64] + impP[tid + 128] + impP[tid + 192];
  // impA is only read after the post-GEMM barrier.

  // ---- phase 2: bf16 WMMA GEMM with register-resident fused reductions ---
  const int mrow = lane & 15;
  const int cOff = (lane < 16) ? 0 : 16;  // contiguous K halves (matches wB)
  const int half = lane >> 4;

  // Per-lane additive partials for this wave's N columns:
  // 10 Gram entries + 4 feat.gws + 4 sum(feat) per (mtile, row-group).
  float red[4][2][18];
#pragma unroll
  for (int mt = 0; mt < 4; ++mt)
#pragma unroll
    for (int gg = 0; gg < 2; ++gg)
#pragma unroll
      for (int j = 0; j < 18; ++j) red[mt][gg][j] = 0.f;

  for (int nb2 = 0; nb2 < 4; ++nb2) {            // 32-column feat blocks
    int ntg0 = wave * 8 + nb2 * 2;               // global 16-col N tiles
    float bv[2], gv[2];
#pragma unroll
    for (int nt = 0; nt < 2; ++nt) {
      int n = (ntg0 + nt) * 16 + mrow;           // lane's C-frag column
      bv[nt] = b_proj[n];
      gv[nt] = gws[n];
    }
    v8f acc[4][2];
#pragma unroll
    for (int mt = 0; mt < 4; ++mt)
#pragma unroll
      for (int nt = 0; nt < 2; ++nt)
        acc[mt][nt] = (v8f){bv[nt], bv[nt], bv[nt], bv[nt],
                            bv[nt], bv[nt], bv[nt], bv[nt]};

    for (int kb = 0; kb < 32; ++kb) {
      int k = kb * 32;
      v16bf afr[4];
#pragma unroll
      for (int mt = 0; mt < 4; ++mt) {
        const bf16* pa = Ab + (size_t)(mt * 16 + mrow) * AP + k + cOff;
        v8bf lo = *(const v8bf*)pa;        // K = k+c .. k+c+7   (ds_load_b128)
        v8bf hi = *(const v8bf*)(pa + 8);  // K = k+c+8 .. k+c+15
        v16bf a;
#pragma unroll
        for (int i = 0; i < 8; ++i) { a[i] = lo[i]; a[i + 8] = hi[i]; }
        afr[mt] = a;
      }
      v16bf bfr[2];
#pragma unroll
      for (int nt = 0; nt < 2; ++nt)
        bfr[nt] = *(const v16bf*)(wB +
                    ((size_t)(ntg0 + nt) * 32 + kb) * 512 + lane * 16);
#pragma unroll
      for (int mt = 0; mt < 4; ++mt)
#pragma unroll
        for (int nt = 0; nt < 2; ++nt)
          acc[mt][nt] = __builtin_amdgcn_wmma_f32_16x16x32_bf16(
              false, afr[mt], false, bfr[nt], (short)0, acc[mt][nt],
              false, false);
    }

    // Lane-local accumulation for this 32-column block (pure VALU FMAs).
    // C-frag layout: lane<16 holds column n, rows 0..7 in v0..7; lane>=16
    // rows 8..15. A chunk is 4 consecutive rows.
#pragma unroll
    for (int mt = 0; mt < 4; ++mt)
#pragma unroll
      for (int nt = 0; nt < 2; ++nt) {
        v8f a = acc[mt][nt];
        float g = gv[nt];
#pragma unroll
        for (int gg = 0; gg < 2; ++gg) {
          float* r = red[mt][gg];
          float r0 = a[gg * 4 + 0], r1 = a[gg * 4 + 1];
          float r2 = a[gg * 4 + 2], r3 = a[gg * 4 + 3];
          r[0] += r0 * r0;  r[1] += r0 * r1;  r[2] += r0 * r2;  r[3] += r0 * r3;
          r[4] += r1 * r1;  r[5] += r1 * r2;  r[6] += r1 * r3;
          r[7] += r2 * r2;  r[8] += r2 * r3;  r[9] += r3 * r3;
          r[10] += r0 * g;  r[11] += r1 * g;  r[12] += r2 * g;  r[13] += r3 * g;
          r[14] += r0;      r[15] += r1;      r[16] += r2;      r[17] += r3;
        }
      }
  }

  // One DPP row-reduction per wave; lanes 15/31 write their half's chunks.
#pragma unroll
  for (int mt = 0; mt < 4; ++mt)
#pragma unroll
    for (int gg = 0; gg < 2; ++gg) {
      int chunk = mt * 4 + half * 2 + gg;
      float* p = part + ((size_t)wave * 16 + chunk) * 18;
#pragma unroll
      for (int j = 0; j < 18; ++j) {
        float v = row_sum16(red[mt][gg][j]);
        if (mrow == 15) p[j] = v;
      }
    }
  __syncthreads();

  // ---- E1: reduce partials across the 8 waves ----------------------------
  for (int q = tid; q < 16 * 18; q += 256) {
    int chunk = q / 18, j = q - chunk * 18;
    float s = 0.f;
#pragma unroll
    for (int wv = 0; wv < 8; ++wv) s += part[((size_t)wv * 16 + chunk) * 18 + j];
    fq[chunk * 18 + j] = s;
  }
  __syncthreads();

  // ---- E2: per-chunk scalar math -> 4 merge coefficients -----------------
  if (tid < 16) {
    const float* q = fq + tid * 18;
    float S[4][4];
    int qi = 0;
    for (int u = 0; u < 4; ++u)
      for (int v = u; v < 4; ++v) { S[u][v] = q[qi]; S[v][u] = q[qi]; ++qi; }
    float sumGws = scal[0], lnbWs = scal[1];
    float nrm[4], smix[4];
    for (int u = 0; u < 4; ++u) {
      nrm[u] = fmaxf(sqrtf(fmaxf(S[u][u], 0.f)), 1e-12f);
      float mu  = q[14 + u] * (1.f / 1024.f);
      float var = S[u][u] * (1.f / 1024.f) - mu * mu;
      float sp  = (q[10 + u] - mu * sumGws) * rsqrtf(var + 1e-5f) + lnbWs;
      float imp = sqrtf(impA[tid * 4 + u]);
      smix[u] = ALPHA_C * sp + (1.f - ALPHA_C) * imp;
    }
    int top = 0;                      // strict > : first-max tie break
    for (int v = 1; v < 4; ++v) if (smix[v] > smix[top]) top = v;
    for (int v = 0; v < 4; ++v) {
      float cf;
      if (v == top) {
        cf = 1.f;
      } else {
        float srow[4];
        for (int j = 0; j < 4; ++j)
          srow[j] = (j == v) ? 0.f : (S[v][j] / (nrm[v] * nrm[j])) * TAU_INV;
        float mx = fmaxf(fmaxf(srow[0], srow[1]), fmaxf(srow[2], srow[3]));
        float e0 = expf(srow[0] - mx), e1 = expf(srow[1] - mx);
        float e2 = expf(srow[2] - mx), e3 = expf(srow[3] - mx);
        float den = e0 + e1 + e2 + e3;
        float et  = (top == 0) ? e0 : (top == 1) ? e1 : (top == 2) ? e2 : e3;
        float p = et / den;           // attn[v][top]
        cf = 0.5f * (p / (p + 1e-6f));// k=1 => norm_w = p/(p+1e-6)
      }
      coefA[tid * 4 + v] = cf;
    }
  }
  __syncthreads();

  // ---- E3: f32 token re-stage via async global->LDS + merged output ------
  for (int h = 0; h < 2; ++h) {
    for (int i = tid; i < 32 * D_DIM; i += 256) {
      int d = i >> 5, t = i & 31;
      unsigned lofs = (unsigned)((t * FP + d) * 4);   // LDS byte offset (Fs@0)
      const float* gp = x + ((size_t)b * D_DIM + d) * T_DIM + t0 + h * 32 + t;
      asm volatile("global_load_async_to_lds_b32 %0, %1, off"
                   :: "v"(lofs), "v"((unsigned long long)(size_t)gp)
                   : "memory");
    }
    asm volatile("s_wait_asynccnt 0x0" ::: "memory");
    __syncthreads();
    for (int i = tid; i < 8 * D_DIM; i += 256) {
      int cl = i & 7, d = i >> 3;
      int chunk = h * 8 + cl;
      const float* cf = coefA + chunk * 4;
      const float* tk = Fs + (size_t)cl * 4 * FP + d;
      float o = cf[0] * tk[0] + cf[1] * tk[FP] +
                cf[2] * tk[2 * FP] + cf[3] * tk[3 * FP];
      out[((size_t)b * D_DIM + d) * 1024 + c0 + chunk] = o;
    }
    __syncthreads();
  }
}

// ---------------------------------------------------------------------------
extern "C" void kernel_launch(void* const* d_in, const int* in_sizes, int n_in,
                              void* d_out, int out_size, void* d_ws, size_t ws_size,
                              hipStream_t stream) {
  (void)in_sizes; (void)n_in; (void)out_size; (void)ws_size;
  const float* x   = (const float*)d_in[0];
  const float* W   = (const float*)d_in[1];
  const float* bp  = (const float*)d_in[2];
  const float* lg  = (const float*)d_in[3];
  const float* lb  = (const float*)d_in[4];
  const float* wsc = (const float*)d_in[5];
  float* out = (float*)d_out;

  bf16*  wB = (bf16*)d_ws;                                  // 2 MB
  float* gw = (float*)((char*)d_ws + 2u * 1024u * 1024u);   // 4 KB
  float* sc = gw + 1024;                                    // 8 B

  repack_w<<<256, 256, 0, stream>>>(W, wB);
  prep_vec<<<1, 256, 0, stream>>>(lg, lb, wsc, gw, sc);

  size_t sh = (size_t)MT_TOK * AP * 2 + (size_t)(8*16*18 + 256 + 64 + 16*18 + 64) * 4;
  fused_tome<<<B_DIM * 64, 256, sh, stream>>>(x, bp, wB, gw, sc, out);
}
